// Encoder_gcl_45913200394643
// MI455X (gfx1250) — compile-verified, matrix operand order checked by reading, not run on hardware
//
#include <hip/hip_runtime.h>

#define NNODES 50000
#define NEDGES 600000
#define CH     128

typedef __attribute__((ext_vector_type(16))) _Float16 v16h;
typedef __attribute__((ext_vector_type(8)))  float    v8f;

// ---------------- degree / norm ----------------
__global__ void init_deg_kernel(float* dis) {
    int i = blockIdx.x * blockDim.x + threadIdx.x;
    if (i < NNODES) dis[i] = 1.0f;               // self-loop
}
__global__ void accum_deg_kernel(const int* __restrict__ dst, float* dis) {
    int e = blockIdx.x * blockDim.x + threadIdx.x;
    if (e < NEDGES) atomicAdd(&dis[dst[e]], 1.0f);
}
__global__ void rsqrt_deg_kernel(float* dis) {
    int i = blockIdx.x * blockDim.x + threadIdx.x;
    if (i < NNODES) dis[i] = rsqrtf(dis[i]);     // deg >= 1 always
}

// ---------------- pack W [128x128 f32 row-major, K x N] into f16 WMMA B-fragment order --
// Fragment f = k_step*8 + col_tile. Lane L holds 16 halfs:
//   n = col_tile*16 + (L&15)
//   elems 0..7 : K = 32*k + 8*(L>>4) + j
//   elems 8..15: K = 32*k + 16 + 8*(L>>4) + j
__global__ void pack_w_kernel(const float* __restrict__ W, _Float16* __restrict__ Wp) {
    int t = blockIdx.x * blockDim.x + threadIdx.x;
    if (t >= CH * CH) return;
    int e = t & 15;
    int L = (t >> 4) & 31;
    int f = t >> 9;              // 0..31
    int k = f >> 3;
    int c = f & 7;
    int K = 32 * k + ((e & 8) ? 16 : 0) + 8 * (L >> 4) + (e & 7);
    int n = c * 16 + (L & 15);
    Wp[t] = (_Float16)W[K * CH + n];
}

// ---------------- GEMM: Out[nrows x 128] = X[nrows x 128] @ W (+bias) via v_wmma_f32_16x16x32_f16
__global__ __launch_bounds__(256) void gemm_x16_kernel(
        const float* __restrict__ X, const _Float16* __restrict__ Wp,
        const float* __restrict__ bias, float* __restrict__ Out, int nrows) {
    const int lane = threadIdx.x & 31;
    const int wave = blockIdx.x * 8 + (threadIdx.x >> 5);
    const int rowBase = wave * 16;
    if (rowBase >= nrows) return;                // wave-uniform: EXEC stays all-ones
    const int half = lane >> 4;
    int m = rowBase + (lane & 15);
    if (m >= nrows) m = nrows - 1;               // select only, no divergence at WMMA
    const float* xrow = X + (size_t)m * CH;

    v8f acc[8] = {};
    #pragma unroll
    for (int k = 0; k < 4; ++k) {
        // A fragment per ISA 16-bit 16x32 layout
        v16h a;
        const float* p0 = xrow + 32 * k + 8 * half;
        const float* p1 = p0 + 16;
        #pragma unroll
        for (int j = 0; j < 8; ++j) {
            a[j]     = (_Float16)p0[j];
            a[j + 8] = (_Float16)p1[j];
        }
        #pragma unroll
        for (int c = 0; c < 8; ++c) {
            v16h b = *(const v16h*)(Wp + (((k * 8 + c) * 32 + lane) << 4));
            acc[c] = __builtin_amdgcn_wmma_f32_16x16x32_f16(
                         false, a, false, b, (short)0, acc[c], false, false);
        }
    }
    // D layout: lanes 0-15: VGPR r -> row rowBase+r, col = c*16+lane; lanes 16-31: row +8
    #pragma unroll
    for (int c = 0; c < 8; ++c) {
        int col = c * 16 + (lane & 15);
        float bv = bias ? bias[col] : 0.0f;
        #pragma unroll
        for (int r = 0; r < 8; ++r) {
            int row = rowBase + r + 8 * half;
            if (row < nrows) Out[(size_t)row * CH + col] = acc[c][r] + bv;
        }
    }
}

// ---------------- edge scatter: wave per edge, lane covers 4 channels ----------------
__global__ __launch_bounds__(256) void scatter_edges_kernel(
        const float* __restrict__ H, const int* __restrict__ src,
        const int* __restrict__ dst, const float* __restrict__ dis,
        float* __restrict__ AGG) {
    int lane = threadIdx.x & 31;
    int e = blockIdx.x * 8 + (threadIdx.x >> 5);
    if (e >= NEDGES) return;
    int s = src[e], d = dst[e];
    float nrm = dis[s] * dis[d];
    const float4 v = *(const float4*)(H + (size_t)s * CH + lane * 4);
    float* out = AGG + (size_t)d * CH + lane * 4;
    atomicAdd(out + 0, v.x * nrm);
    atomicAdd(out + 1, v.y * nrm);
    atomicAdd(out + 2, v.z * nrm);
    atomicAdd(out + 3, v.w * nrm);
}

// ---------------- epilogues: self-loop + bias + PReLU (+ skip accumulate / final store) --
__global__ void epilogue_add_kernel(const float* __restrict__ AGG, const float* __restrict__ H,
                                    const float* __restrict__ dis, const float* __restrict__ b,
                                    const float* __restrict__ a, float* __restrict__ U) {
    int idx = blockIdx.x * blockDim.x + threadIdx.x;
    if (idx >= NNODES * CH) return;
    int i = idx >> 7, c = idx & 127;
    float di = dis[i];
    float v = AGG[idx] + di * di * H[idx] + b[c];
    float h = v >= 0.0f ? v : a[c] * v;
    U[idx] += h;                                  // skip-sum: U already holds x@Ws + bs
}
__global__ void epilogue_out_kernel(const float* __restrict__ AGG, const float* __restrict__ H,
                                    const float* __restrict__ dis, const float* __restrict__ b,
                                    const float* __restrict__ a, float* __restrict__ out) {
    int idx = blockIdx.x * blockDim.x + threadIdx.x;
    if (idx >= NNODES * CH) return;
    int i = idx >> 7, c = idx & 127;
    float di = dis[i];
    float v = AGG[idx] + di * di * H[idx] + b[c];
    out[idx] = v >= 0.0f ? v : a[c] * v;
}

extern "C" void kernel_launch(void* const* d_in, const int* in_sizes, int n_in,
                              void* d_out, int out_size, void* d_ws, size_t ws_size,
                              hipStream_t stream) {
    const float* x  = (const float*)d_in[0];
    const int* edge = (const int*)d_in[1];
    const int* src  = edge;
    const int* dst  = edge + NEDGES;
    const float* W0 = (const float*)d_in[2];
    const float* b0 = (const float*)d_in[3];
    const float* W1 = (const float*)d_in[4];
    const float* b1 = (const float*)d_in[5];
    const float* Ws = (const float*)d_in[6];
    const float* bs = (const float*)d_in[7];
    const float* a  = (const float*)d_in[8];
    float* out = (float*)d_out;

    // workspace layout (~77 MB)
    float* dis = (float*)d_ws;
    float* H   = dis + 50176;
    float* AGG = H + (size_t)NNODES * CH;
    float* U   = AGG + (size_t)NNODES * CH;
    _Float16* Wp0 = (_Float16*)(U + (size_t)NNODES * CH);
    _Float16* Wps = Wp0 + CH * CH;
    _Float16* Wp1 = Wps + CH * CH;

    const int nodeBlocks = (NNODES + 255) / 256;
    const int edgeBlocks = (NEDGES + 255) / 256;
    const int elemBlocks = (NNODES * CH + 255) / 256;
    const int gemmBlocks = (NNODES + 127) / 128;          // 8 waves/block, 16 rows/wave
    const int scatBlocks = (NEDGES + 7) / 8;              // wave per edge

    // norms
    init_deg_kernel<<<nodeBlocks, 256, 0, stream>>>(dis);
    accum_deg_kernel<<<edgeBlocks, 256, 0, stream>>>(dst, dis);
    rsqrt_deg_kernel<<<nodeBlocks, 256, 0, stream>>>(dis);

    // pack weights to f16 fragment order
    pack_w_kernel<<<64, 256, 0, stream>>>(W0, Wp0);
    pack_w_kernel<<<64, 256, 0, stream>>>(Ws, Wps);
    pack_w_kernel<<<64, 256, 0, stream>>>(W1, Wp1);

    // layer 0: H = x@W0 ; U = x@Ws + bs
    gemm_x16_kernel<<<gemmBlocks, 256, 0, stream>>>(x, Wp0, nullptr, H, NNODES);
    gemm_x16_kernel<<<gemmBlocks, 256, 0, stream>>>(x, Wps, bs, U, NNODES);
    hipMemsetAsync(AGG, 0, (size_t)NNODES * CH * sizeof(float), stream);
    scatter_edges_kernel<<<scatBlocks, 256, 0, stream>>>(H, src, dst, dis, AGG);
    epilogue_add_kernel<<<elemBlocks, 256, 0, stream>>>(AGG, H, dis, b0, a, U);

    // layer 1: H = U@W1 ; aggregate ; output
    gemm_x16_kernel<<<gemmBlocks, 256, 0, stream>>>(U, Wp1, nullptr, H, NNODES);
    hipMemsetAsync(AGG, 0, (size_t)NNODES * CH * sizeof(float), stream);
    scatter_edges_kernel<<<scatBlocks, 256, 0, stream>>>(H, src, dst, dis, AGG);
    epilogue_out_kernel<<<elemBlocks, 256, 0, stream>>>(AGG, H, dis, b1, a, out);
}
